// NSMOE_61649960566995
// MI455X (gfx1250) — matrix operand order repaired and checked by reading, not compiled, not run on hardware
//
#include <hip/hip_runtime.h>

#define TOK   8192          // B*S tokens
#define DIM   768
#define NEXP  8
#define TOPK  2
#define NENT  (TOK*TOPK)    // routed entries
#define KSTEPS (DIM/32)     // 24

// CDNA5 async global->LDS path (ASYNCcnt-tracked), if the toolchain exposes it.
#if defined(__has_builtin)
#  if __has_builtin(__builtin_amdgcn_global_load_async_to_lds_b128) && \
      __has_builtin(__builtin_amdgcn_s_wait_asynccnt)
#    define USE_ASYNC 1
#  endif
#endif
#ifndef USE_ASYNC
#  define USE_ASYNC 0
#endif

typedef __attribute__((ext_vector_type(16))) __bf16 v16bf;
typedef __attribute__((ext_vector_type(8)))  float  v8f;
typedef __attribute__((ext_vector_type(4)))  int    v4i;

union ABu { uint4 u[2]; v16bf v; };

__device__ __forceinline__ unsigned short f2bf(float f) {
  unsigned u = __float_as_uint(f);
  u += 0x7FFFu + ((u >> 16) & 1u);          // round-to-nearest-even
  return (unsigned short)(u >> 16);
}
__device__ __forceinline__ float bf2f(unsigned short h) {
  return __uint_as_float(((unsigned)h) << 16);
}

// ---------------------------------------------------------------- conversions
__global__ __launch_bounds__(256) void k_cvt_x(const float* __restrict__ x,
                                               unsigned short* __restrict__ xbf) {
  int i = blockIdx.x * 256 + threadIdx.x;
  if (i < TOK * DIM) xbf[i] = f2bf(x[i]);
}

// wt[e][n][k] = w[e][k][n] -- LDS-tiled 32x32 transpose, coalesced both ways.
__global__ __launch_bounds__(256) void k_cvt_wT(const float* __restrict__ w,
                                                unsigned short* __restrict__ wt) {
  __shared__ unsigned short tile[32][34];
  const int e  = blockIdx.z;
  const int k0 = blockIdx.x * 32;
  const int n0 = blockIdx.y * 32;
  const int tx = threadIdx.x & 31;
  const int ty = threadIdx.x >> 5;              // 0..7
  const float* src = w + (size_t)e * DIM * DIM;
  #pragma unroll
  for (int i = 0; i < 4; ++i) {                 // read w[k][n], n contiguous
    int k = k0 + ty + i * 8;
    tile[ty + i * 8][tx] = f2bf(src[(size_t)k * DIM + n0 + tx]);
  }
  __syncthreads();
  unsigned short* dst = wt + (size_t)e * DIM * DIM;
  #pragma unroll
  for (int i = 0; i < 4; ++i) {                 // write wt[n][k], k contiguous
    int n = n0 + ty + i * 8;
    dst[(size_t)n * DIM + k0 + tx] = tile[tx][ty + i * 8];
  }
}

// ---------------------------------------------------------------- gating
__global__ __launch_bounds__(256) void k_gate(const float* __restrict__ x,
                                              const float* __restrict__ gw,
                                              const float* __restrict__ gb,
                                              int*   __restrict__ topi,
                                              float* __restrict__ rw,
                                              float* __restrict__ scores) {
  const int lane = threadIdx.x & 31;
  const int wid  = threadIdx.x >> 5;
  const int t    = blockIdx.x * 8 + wid;        // one wave32 per token
  const float* xr = x + (size_t)t * DIM;

  float p0=0,p1=0,p2=0,p3=0,p4=0,p5=0,p6=0,p7=0;
  for (int d = lane; d < DIM; d += 32) {
    float xv = xr[d];
    const float4* g = (const float4*)(gw + d * NEXP);
    float4 ga = g[0], gc = g[1];
    p0 += xv * ga.x; p1 += xv * ga.y; p2 += xv * ga.z; p3 += xv * ga.w;
    p4 += xv * gc.x; p5 += xv * gc.y; p6 += xv * gc.z; p7 += xv * gc.w;
  }
  #pragma unroll
  for (int off = 16; off; off >>= 1) {
    p0 += __shfl_down(p0, off, 32); p1 += __shfl_down(p1, off, 32);
    p2 += __shfl_down(p2, off, 32); p3 += __shfl_down(p3, off, 32);
    p4 += __shfl_down(p4, off, 32); p5 += __shfl_down(p5, off, 32);
    p6 += __shfl_down(p6, off, 32); p7 += __shfl_down(p7, off, 32);
  }
  if (lane == 0) {
    float l[NEXP] = { p0+gb[0], p1+gb[1], p2+gb[2], p3+gb[3],
                      p4+gb[4], p5+gb[5], p6+gb[6], p7+gb[7] };
    float mx = l[0];
    #pragma unroll
    for (int e = 1; e < NEXP; ++e) mx = fmaxf(mx, l[e]);
    float sc[NEXP], s = 0.f;
    #pragma unroll
    for (int e = 0; e < NEXP; ++e) { sc[e] = __expf(l[e] - mx); s += sc[e]; }
    float inv = 1.f / s;
    #pragma unroll
    for (int e = 0; e < NEXP; ++e) { sc[e] *= inv; scores[t * NEXP + e] = sc[e]; }
    int i0 = 0;
    #pragma unroll
    for (int e = 1; e < NEXP; ++e) if (sc[e] > sc[i0]) i0 = e;
    int i1 = (i0 == 0) ? 1 : 0;
    #pragma unroll
    for (int e = 0; e < NEXP; ++e) if (e != i0 && sc[e] > sc[i1]) i1 = e;
    float den = sc[i0] + sc[i1] + 1e-8f;
    topi[2 * t]     = i0;  rw[2 * t]     = sc[i0] / den;
    topi[2 * t + 1] = i1;  rw[2 * t + 1] = sc[i1] / den;
  }
}

// ---------------------------------------------------------------- routing
__global__ void k_zero_cnt(int* cnt) { if (threadIdx.x < NEXP) cnt[threadIdx.x] = 0; }

__global__ __launch_bounds__(256) void k_route(const int* __restrict__ topi,
                                               int* __restrict__ cnt,
                                               int* __restrict__ list) {
  int i = blockIdx.x * 256 + threadIdx.x;       // entry id = token*2 + k
  if (i < NENT) {
    int e = topi[i];
    int slot = atomicAdd(&cnt[e], 1);
    list[e * TOK + slot] = i;
  }
}

// ---------------------------------------------------------------- expert GEMM
// MODE 1: h = relu(gather(xbf) @ w1^T_tile + b1)   (A row = token = entry>>1)
// MODE 2: y = gather(h) @ w2^T_tile + b2           (A row = entry)
template <int MODE>
__global__ __launch_bounds__(256) void k_ffn(const unsigned short* __restrict__ Abase,
                                             const unsigned short* __restrict__ Wbf,
                                             const float* __restrict__ bias,
                                             const int* __restrict__ list,
                                             const int* __restrict__ cnt,
                                             unsigned short* __restrict__ Obf) {
  const int e     = blockIdx.z;
  const int mtile = blockIdx.x;
  const int nbase = blockIdx.y * 128;
  const int count = cnt[e];
  if (mtile * 128 >= count) return;             // uniform exit: EXEC stays all-1

  __shared__ unsigned short sA[2][128 * 32];    // double-buffered A tile (bf16)
  __shared__ unsigned short sB[2][128 * 32];    // double-buffered W tile (bf16)
  __shared__ int sIdx[128];

  const int tid  = threadIdx.x;
  const int lane = tid & 31;
  const int wid  = tid >> 5;

  if (tid < 128) {
    int m = mtile * 128 + tid;
    sIdx[tid] = list[e * TOK + (m < count ? m : 0)];
  }
  __syncthreads();

  const int row  = tid >> 1;                    // 0..127
  const int half = tid & 1;                     // which 16-elem half of K-step
  int ent = sIdx[row];
  size_t srcRow = (MODE == 1) ? (size_t)(ent >> 1) : (size_t)ent;
  const unsigned short* aSrc = Abase + srcRow * DIM;
  const unsigned short* bSrc = Wbf + ((size_t)e * DIM + (nbase + row)) * DIM;

  v8f acc[8];
  #pragma unroll
  for (int a = 0; a < 8; ++a) acc[a] = (v8f){0,0,0,0,0,0,0,0};

  const int ak0  = (lane & 16) ? 16 : 0;        // A frag: K 0..15 | 16..31
  const int bk0  = (lane & 16) ? 16 : 0;        // B frag: K 0..15 | 16..31
  const int arow = wid * 16 + (lane & 15);
  const int ldsOff = row * 32 + half * 16;      // this thread's 32B slot in a tile

#if USE_ASYNC
  typedef v4i __attribute__((address_space(1)))* gv4p;   // global v4i32*
  typedef v4i __attribute__((address_space(3)))* lv4p;   // LDS v4i32*
  // ---- async global->LDS, double buffered: 4 ASYNCcnt ops per stage ----
  #define STAGE(KT, BUF)                                                       \
    do {                                                                       \
      const unsigned short* ga  = aSrc + (KT) * 32 + half * 16;                \
      const unsigned short* gb_ = bSrc + (KT) * 32 + half * 16;                \
      __builtin_amdgcn_global_load_async_to_lds_b128(                          \
          (gv4p)(void*)ga,        (lv4p)(void*)&sA[BUF][ldsOff], 0, 0);        \
      __builtin_amdgcn_global_load_async_to_lds_b128(                          \
          (gv4p)(void*)(ga + 8),  (lv4p)(void*)&sA[BUF][ldsOff + 8], 0, 0);    \
      __builtin_amdgcn_global_load_async_to_lds_b128(                          \
          (gv4p)(void*)gb_,       (lv4p)(void*)&sB[BUF][ldsOff], 0, 0);        \
      __builtin_amdgcn_global_load_async_to_lds_b128(                          \
          (gv4p)(void*)(gb_ + 8), (lv4p)(void*)&sB[BUF][ldsOff + 8], 0, 0);    \
    } while (0)

  STAGE(0, 0);
  for (int kt = 0; kt < KSTEPS; ++kt) {
    const int cur = kt & 1;
    if (kt + 1 < KSTEPS) {
      STAGE(kt + 1, cur ^ 1);                   // prefetch next tile into other buf
      __builtin_amdgcn_s_wait_asynccnt(4);      // older 4 ops (current tile) done
    } else {
      __builtin_amdgcn_s_wait_asynccnt(0);
    }
    __syncthreads();

    ABu af;
    af.u[0] = *(const uint4*)&sA[cur][arow * 32 + ak0];
    af.u[1] = *(const uint4*)&sA[cur][arow * 32 + ak0 + 8];
    #pragma unroll
    for (int a = 0; a < 8; ++a) {
      int nl = a * 16 + (lane & 15);
      ABu bfr;
      bfr.u[0] = *(const uint4*)&sB[cur][nl * 32 + bk0];
      bfr.u[1] = *(const uint4*)&sB[cur][nl * 32 + bk0 + 8];
      acc[a] = __builtin_amdgcn_wmma_f32_16x16x32_bf16(
          false, af.v, false, bfr.v, (short)0, acc[a], false, false);
    }
    __syncthreads();
  }
  #undef STAGE
#else
  // ---- fallback: synchronous global->VGPR->LDS staging ----
  for (int kt = 0; kt < KSTEPS; ++kt) {
    const uint4* ap = (const uint4*)(aSrc + kt * 32 + half * 16);
    const uint4* bp = (const uint4*)(bSrc + kt * 32 + half * 16);
    uint4 a0 = ap[0], a1 = ap[1];
    uint4 b0 = bp[0], b1v = bp[1];
    if (kt + 1 < KSTEPS) {
      __builtin_prefetch((const void*)(aSrc + (kt + 1) * 32), 0, 0);
      __builtin_prefetch((const void*)(bSrc + (kt + 1) * 32), 0, 0);
    }
    uint4* as = (uint4*)&sA[0][ldsOff];
    uint4* bs = (uint4*)&sB[0][ldsOff];
    as[0] = a0; as[1] = a1;
    bs[0] = b0; bs[1] = b1v;
    __syncthreads();

    ABu af;
    af.u[0] = *(const uint4*)&sA[0][arow * 32 + ak0];
    af.u[1] = *(const uint4*)&sA[0][arow * 32 + ak0 + 8];
    #pragma unroll
    for (int a = 0; a < 8; ++a) {
      int nl = a * 16 + (lane & 15);
      ABu bfr;
      bfr.u[0] = *(const uint4*)&sB[0][nl * 32 + bk0];
      bfr.u[1] = *(const uint4*)&sB[0][nl * 32 + bk0 + 8];
      acc[a] = __builtin_amdgcn_wmma_f32_16x16x32_bf16(
          false, af.v, false, bfr.v, (short)0, acc[a], false, false);
    }
    __syncthreads();
  }
#endif

  // epilogue: bias (+relu) -> bf16, scatter by entry id
  #pragma unroll
  for (int a = 0; a < 8; ++a) {
    int n = nbase + a * 16 + (lane & 15);
    float bi = bias[e * DIM + n];
    #pragma unroll
    for (int r = 0; r < 8; ++r) {
      int mrow = wid * 16 + ((lane & 16) ? r + 8 : r);
      if (mtile * 128 + mrow < count) {
        int en = sIdx[mrow];
        float v = acc[a][r] + bi;
        if (MODE == 1) v = fmaxf(v, 0.f);
        Obf[(size_t)en * DIM + n] = f2bf(v);
      }
    }
  }
}

// ---------------------------------------------------------------- combine
__global__ __launch_bounds__(256) void k_combine(const unsigned short* __restrict__ y,
                                                 const float* __restrict__ rw,
                                                 float* __restrict__ out) {
  int i = blockIdx.x * 256 + threadIdx.x;
  if (i < TOK * DIM) {
    int t = i / DIM, n = i % DIM;
    float v0 = bf2f(y[(size_t)(2 * t) * DIM + n])     * rw[2 * t];
    float v1 = bf2f(y[(size_t)(2 * t + 1) * DIM + n]) * rw[2 * t + 1];
    out[i] = v0 + v1;
  }
}

// ---------------------------------------------------------------- aux loss
__global__ __launch_bounds__(256) void k_aux(const float* __restrict__ scores,
                                             float* __restrict__ out_aux) {
  __shared__ float s[256][NEXP];
  float p[NEXP] = {0,0,0,0,0,0,0,0};
  for (int t = threadIdx.x; t < TOK; t += 256) {
    #pragma unroll
    for (int e = 0; e < NEXP; ++e) p[e] += scores[t * NEXP + e];
  }
  #pragma unroll
  for (int e = 0; e < NEXP; ++e) s[threadIdx.x][e] = p[e];
  __syncthreads();
  for (int off = 128; off; off >>= 1) {
    if (threadIdx.x < off) {
      #pragma unroll
      for (int e = 0; e < NEXP; ++e) s[threadIdx.x][e] += s[threadIdx.x + off][e];
    }
    __syncthreads();
  }
  if (threadIdx.x == 0) {
    float mean = 0.f;
    #pragma unroll
    for (int e = 0; e < NEXP; ++e) mean += s[0][e];
    mean *= (1.f / NEXP);
    float var = 0.f;
    #pragma unroll
    for (int e = 0; e < NEXP; ++e) { float d = s[0][e] - mean; var += d * d; }
    var *= (1.f / NEXP);
    out_aux[0] = 0.01f * var / (mean * mean + 1e-8f);
  }
}

// ---------------------------------------------------------------- launcher
extern "C" void kernel_launch(void* const* d_in, const int* in_sizes, int n_in,
                              void* d_out, int out_size, void* d_ws, size_t ws_size,
                              hipStream_t stream) {
  const float* x  = (const float*)d_in[0];
  const float* gw = (const float*)d_in[1];
  const float* gb = (const float*)d_in[2];
  const float* w1 = (const float*)d_in[3];
  const float* b1 = (const float*)d_in[4];
  const float* w2 = (const float*)d_in[5];
  const float* b2 = (const float*)d_in[6];
  float* out = (float*)d_out;                   // [TOK*DIM] out, then [1] aux

  // workspace layout (256B-aligned)
  char* ws = (char*)d_ws;
  auto align = [](size_t v) { return (v + 255) & ~(size_t)255; };
  size_t off = 0;
  unsigned short* xbf = (unsigned short*)(ws + off); off = align(off + (size_t)TOK * DIM * 2);
  unsigned short* wt1 = (unsigned short*)(ws + off); off = align(off + (size_t)NEXP * DIM * DIM * 2);
  unsigned short* wt2 = (unsigned short*)(ws + off); off = align(off + (size_t)NEXP * DIM * DIM * 2);
  unsigned short* h   = (unsigned short*)(ws + off); off = align(off + (size_t)NENT * DIM * 2);
  unsigned short* y   = (unsigned short*)(ws + off); off = align(off + (size_t)NENT * DIM * 2);
  float* scores = (float*)(ws + off); off = align(off + (size_t)TOK * NEXP * 4);
  float* rw     = (float*)(ws + off); off = align(off + (size_t)NENT * 4);
  int*   topi   = (int*)(ws + off);   off = align(off + (size_t)NENT * 4);
  int*   cnt    = (int*)(ws + off);   off = align(off + 256);
  int*   list   = (int*)(ws + off);   off = align(off + (size_t)NEXP * TOK * 4);
  (void)ws_size; (void)in_sizes; (void)n_in; (void)out_size;

  const int NXD = TOK * DIM;                    // 6,291,456

  k_cvt_x<<<(NXD + 255) / 256, 256, 0, stream>>>(x, xbf);
  dim3 gT(DIM / 32, DIM / 32, NEXP);            // (24, 24, 8)
  k_cvt_wT<<<gT, 256, 0, stream>>>(w1, wt1);
  k_cvt_wT<<<gT, 256, 0, stream>>>(w2, wt2);
  k_zero_cnt<<<1, 32, 0, stream>>>(cnt);
  k_gate  <<<TOK / 8, 256, 0, stream>>>(x, gw, gb, topi, rw, scores);
  k_route <<<(NENT + 255) / 256, 256, 0, stream>>>(topi, cnt, list);

  dim3 gFFN(TOK / 128, DIM / 128, NEXP);        // (64, 6, 8)
  k_ffn<1><<<gFFN, 256, 0, stream>>>(xbf, wt1, b1, list, cnt, h);
  k_ffn<2><<<gFFN, 256, 0, stream>>>(h,   wt2, b2, list, cnt, y);

  k_combine<<<(NXD + 255) / 256, 256, 0, stream>>>(y, rw, out);
  k_aux    <<<1, 256, 0, stream>>>(scores, out + NXD);
}
